// S2TDPSA_84988812853846
// MI455X (gfx1250) — matrix operand order repaired and checked by reading, not compiled, not run on hardware
//
#include <hip/hip_runtime.h>
#include <hip/hip_bf16.h>
#include <math.h>

// ---------------------------------------------------------------------------
// S2TDPSA for MI455X (gfx1250): wave32 + v_wmma_f32_16x16x32_bf16 everywhere.
// A-matrix of the attention is rank-factored per sign branch and generated in
// registers (never touches HBM). All GEMMs are LDS-tiled bf16 WMMA.
// ---------------------------------------------------------------------------

typedef __bf16 bf16_t;
typedef __bf16 v16bf __attribute__((ext_vector_type(16)));
typedef float  v8f   __attribute__((ext_vector_type(8)));

#define B_    8
#define N_    2048
#define D_    1024
#define EPS_  1e-5f
#define ASTDP 0.9f
#define INV_TAU 2.0f   // 1/0.5
#define WOFF  0.9f

#define BM 128
#define BN 64
#define KB 32
#define ASTRIDE 40   // 80B rows: 16B aligned, 20-dword bank stride (conflict free)
#define BSTRIDE 72   // 144B rows: 16B aligned, 36-dword bank stride

__device__ __forceinline__ float sigmoid4(float h) {
  // sigmoid(4*(h - V_TH)), V_TH = 1
  return 1.0f / (1.0f + __expf(-4.0f * (h - 1.0f)));
}

// ---------------- small helper kernels ----------------

__global__ void cvt_bf16_kernel(const float* __restrict__ in, bf16_t* __restrict__ out, int n) {
  int i = blockIdx.x * 256 + threadIdx.x;
  if (i < n) out[i] = (bf16_t)in[i];
}

// W -> bf16 with per-output-column BN scale folded in (g * rsqrt(v+eps)); g==null => scale 1
__global__ void cvt_w_kernel(const float* __restrict__ W, const float* __restrict__ g,
                             const float* __restrict__ vv, bf16_t* __restrict__ out) {
  int i = blockIdx.x * 256 + threadIdx.x;           // D*D
  int d = i & (D_ - 1);
  float sc = g ? g[d] * rsqrtf(vv[d] + EPS_) : 1.0f;
  out[i] = (bf16_t)(W[i] * sc);
}

__global__ void shift_kernel(const float* __restrict__ g, const float* __restrict__ bb,
                             const float* __restrict__ m, const float* __restrict__ vv,
                             float* __restrict__ sh) {
  int d = blockIdx.x * 256 + threadIdx.x;
  float sc = g[d] * rsqrtf(vv[d] + EPS_);
  sh[d] = bb[d] - m[d] * sc;
}

// tq = 1 - rowsum/D ; precompute sign-branch exponential factors (A_STDP folded into aq/bq)
__global__ void finish_t_kernel(const float* __restrict__ rq, const float* __restrict__ rk,
                                float* __restrict__ tq, float* __restrict__ aq, float* __restrict__ bq,
                                float* __restrict__ tk, float* __restrict__ ck, float* __restrict__ dk) {
  int i = blockIdx.x * 256 + threadIdx.x;           // B*N
  float tQ = 1.0f - rq[i] * (1.0f / (float)D_);
  float tK = 1.0f - rk[i] * (1.0f / (float)D_);
  tq[i] = tQ; tk[i] = tK;
  aq[i] = ASTDP * __expf( tQ * INV_TAU);
  bq[i] = ASTDP * __expf(-tQ * INV_TAU);
  ck[i] = __expf(-tK * INV_TAU);
  dk[i] = __expf( tK * INV_TAU);
}

// colV[b,d] = W_OFF * sum_n V[b,n,d]  (8 n-chunks of 256, atomics; colV pre-zeroed)
__global__ void colsum_v_kernel(const bf16_t* __restrict__ Vb, float* __restrict__ colV) {
  int t = blockIdx.x * 256 + threadIdx.x;           // B*D*8
  int b     = t / (D_ * 8);
  int rem   = t % (D_ * 8);
  int chunk = rem / D_;
  int d     = rem % D_;
  const bf16_t* p = Vb + (size_t)b * N_ * D_ + (size_t)(chunk * 256) * D_ + d;
  float s = 0.0f;
  for (int n = 0; n < 256; ++n) s += (float)p[(size_t)n * D_];
  atomicAdd(&colV[b * D_ + d], WOFF * s);
}

// ---------------- tiled bf16 WMMA GEMM ----------------
// MODE 0: epilogue sigmoid4(acc+shift[col]) reduced to per-row sums (atomicAdd into outf[M])
// MODE 1: epilogue sigmoid4(acc+shift[col]) stored bf16 to outb[M,NC]
// MODE 2: epilogue acc + shift[col] (bias) stored f32 to outf[M,NC]
template <int MODE>
__global__ __launch_bounds__(256)
void gemm_bf16(const bf16_t* __restrict__ Am, const bf16_t* __restrict__ Bm,
               const float* __restrict__ shift,
               float* __restrict__ outf, bf16_t* __restrict__ outb,
               int M, int K, int NC) {
  __shared__ bf16_t As[BM * ASTRIDE];
  __shared__ bf16_t Bs[KB * BSTRIDE];

  const int tid  = threadIdx.x;
  const int lane = tid & 31;
  const int wave = tid >> 5;
  const int wm = wave >> 1, wn = wave & 1;
  const int rowBase = blockIdx.y * BM;
  const int nBase   = blockIdx.x * BN;
  const int half8 = (lane < 16) ? 0 : 8;   // A-frag K base per lane-half (ISA layout)
  const int kAdd  = (lane < 16) ? 0 : 16;  // unused; B frag uses half8 as its K base

  v8f acc[2][2] = {};

  for (int k0 = 0; k0 < K; k0 += KB) {
    // stage A tile 128x32 (512 16B chunks, 2/thread)
#pragma unroll
    for (int c = 0; c < 2; ++c) {
      int ci = tid + c * 256;
      int r = ci >> 2, cc = ci & 3;
      *(uint4*)(&As[r * ASTRIDE + cc * 8]) =
          *(const uint4*)(Am + (size_t)(rowBase + r) * K + k0 + cc * 8);
    }
    // stage B tile 32x64 (256 16B chunks, 1/thread)
    {
      int r = tid >> 3, cc = tid & 7;
      *(uint4*)(&Bs[r * BSTRIDE + cc * 8]) =
          *(const uint4*)(Bm + (size_t)(k0 + r) * NC + nBase + cc * 8);
    }
    if (k0 + KB < K) {  // hint next tiles toward L2 (global_prefetch)
      __builtin_prefetch(Am + (size_t)(rowBase + (tid >> 2)) * K + k0 + KB, 0, 3);
      __builtin_prefetch(Bm + (size_t)(k0 + KB + (tid >> 3)) * NC + nBase, 0, 3);
    }
    __syncthreads();

    v16bf af[2], bfv[2];
#pragma unroll
    for (int mt = 0; mt < 2; ++mt) {
      // lane holds row M=lane%16; K runs {base..base+7, base+16..base+23}
      const bf16_t* rp = &As[(wm * 32 + mt * 16 + (lane & 15)) * ASTRIDE + half8];
      uint2 p0 = *(const uint2*)(rp);
      uint2 p1 = *(const uint2*)(rp + 4);
      uint2 p2 = *(const uint2*)(rp + 16);
      uint2 p3 = *(const uint2*)(rp + 20);
      uint2* dst = (uint2*)&af[mt];
      dst[0] = p0; dst[1] = p1; dst[2] = p2; dst[3] = p3;
    }
#pragma unroll
    for (int nt = 0; nt < 2; ++nt) {
      int n = wn * 32 + nt * 16 + (lane & 15);
#pragma unroll
      for (int e = 0; e < 16; ++e)
        bfv[nt][e] = Bs[(half8 + e + ((e >= 8) ? 8 : 0)) * BSTRIDE + n];
    }
#pragma unroll
    for (int mt = 0; mt < 2; ++mt)
#pragma unroll
      for (int nt = 0; nt < 2; ++nt)
        acc[mt][nt] = __builtin_amdgcn_wmma_f32_16x16x32_bf16(
            false, af[mt], false, bfv[nt], (short)0, acc[mt][nt], false, false);
    __syncthreads();
  }

  // epilogue: C layout -> VGPR r: lanes0-15 row r (N=lane), lanes16-31 row r+8
#pragma unroll
  for (int mt = 0; mt < 2; ++mt) {
#pragma unroll
    for (int nt = 0; nt < 2; ++nt) {
      int col  = nBase + wn * 32 + nt * 16 + (lane & 15);
      int row0 = rowBase + wm * 32 + mt * 16 + ((lane < 16) ? 0 : 8);
      float sh = shift[col];
      if (MODE == 0) {
#pragma unroll
        for (int r = 0; r < 8; ++r) {
          float s = sigmoid4(acc[mt][nt][r] + sh);
          s += __shfl_xor(s, 1, 32);
          s += __shfl_xor(s, 2, 32);
          s += __shfl_xor(s, 4, 32);
          s += __shfl_xor(s, 8, 32);
          if ((lane & 15) == 0) atomicAdd(&outf[row0 + r], s);
        }
      } else if (MODE == 1) {
#pragma unroll
        for (int r = 0; r < 8; ++r)
          outb[(size_t)(row0 + r) * NC + col] = (bf16_t)sigmoid4(acc[mt][nt][r] + sh);
      } else {
#pragma unroll
        for (int r = 0; r < 8; ++r)
          outf[(size_t)(row0 + r) * NC + col] = acc[mt][nt][r] + sh;
      }
    }
  }
}

// ---------------- attention GEMM: O1 = F@V + W_OFF*colsum(V) ----------------
// F[i,j] = (tq_i < tk_j) ? aq_i*ck_j : -bq_i*dk_j  (A_STDP folded into aq/bq),
// generated directly into WMMA A fragments (no HBM traffic for the NxN matrix).
// All per-element selects are branchless (both products computed, v_cndmask),
// and the per-lane j-factors are fetched with ds_load_b128 (two contiguous
// 8-float runs per lane: [jAdd,jAdd+8) and [jAdd+16,jAdd+24)).
__global__ __launch_bounds__(256)
void attn_gemm(const bf16_t* __restrict__ Vb,
               const float* __restrict__ tq, const float* __restrict__ aq, const float* __restrict__ bq,
               const float* __restrict__ tk, const float* __restrict__ ck, const float* __restrict__ dk,
               const float* __restrict__ colV, bf16_t* __restrict__ O1) {
  __shared__ bf16_t Bs[KB * BSTRIDE];
  __shared__ __align__(16) float tkS[KB];
  __shared__ __align__(16) float ckS[KB];
  __shared__ __align__(16) float dkS[KB];

  const int b = blockIdx.z;
  const bf16_t* V = Vb + (size_t)b * N_ * D_;
  const float* tqB = tq + b * N_;
  const float* aqB = aq + b * N_;
  const float* bqB = bq + b * N_;
  const float* tkB = tk + b * N_;
  const float* ckB = ck + b * N_;
  const float* dkB = dk + b * N_;

  const int tid  = threadIdx.x;
  const int lane = tid & 31;
  const int wave = tid >> 5;
  const int wm = wave >> 1, wn = wave & 1;
  const int rowBase = blockIdx.y * BM;
  const int nBase   = blockIdx.x * BN;
  const int kAdd = (lane < 16) ? 0 : 16;
  const int jAdd = (lane < 16) ? 0 : 8;

  float tqv[2], aqv[2], bqv[2];
#pragma unroll
  for (int mt = 0; mt < 2; ++mt) {
    int i = rowBase + wm * 32 + mt * 16 + (lane & 15);
    tqv[mt] = tqB[i]; aqv[mt] = aqB[i]; bqv[mt] = bqB[i];
  }

  v8f acc[2][2] = {};

  for (int k0 = 0; k0 < N_; k0 += KB) {
    if (tid < KB) { tkS[tid] = tkB[k0 + tid]; ckS[tid] = ckB[k0 + tid]; dkS[tid] = dkB[k0 + tid]; }
    {
      int r = tid >> 3, cc = tid & 7;
      *(uint4*)(&Bs[r * BSTRIDE + cc * 8]) =
          *(const uint4*)(V + (size_t)(k0 + r) * D_ + nBase + cc * 8);
    }
    __syncthreads();

    // fetch this lane's 16 (tk,ck,dk) triples with vector LDS loads
    float tr[16], cr[16], dr[16];
    *(float4*)&tr[0]  = *(const float4*)&tkS[jAdd];
    *(float4*)&tr[4]  = *(const float4*)&tkS[jAdd + 4];
    *(float4*)&tr[8]  = *(const float4*)&tkS[jAdd + 16];
    *(float4*)&tr[12] = *(const float4*)&tkS[jAdd + 20];
    *(float4*)&cr[0]  = *(const float4*)&ckS[jAdd];
    *(float4*)&cr[4]  = *(const float4*)&ckS[jAdd + 4];
    *(float4*)&cr[8]  = *(const float4*)&ckS[jAdd + 16];
    *(float4*)&cr[12] = *(const float4*)&ckS[jAdd + 20];
    *(float4*)&dr[0]  = *(const float4*)&dkS[jAdd];
    *(float4*)&dr[4]  = *(const float4*)&dkS[jAdd + 4];
    *(float4*)&dr[8]  = *(const float4*)&dkS[jAdd + 16];
    *(float4*)&dr[12] = *(const float4*)&dkS[jAdd + 20];

    v16bf af[2], bfv[2];
#pragma unroll
    for (int mt = 0; mt < 2; ++mt) {
#pragma unroll
      for (int e = 0; e < 16; ++e) {
        float fpos = aqv[mt] * cr[e];        // dt<0 branch
        float fneg = -bqv[mt] * dr[e];       // dt>=0 branch
        af[mt][e] = (bf16_t)((tqv[mt] < tr[e]) ? fpos : fneg);  // v_cndmask
      }
    }
#pragma unroll
    for (int nt = 0; nt < 2; ++nt) {
      int n = wn * 32 + nt * 16 + (lane & 15);
#pragma unroll
      for (int e = 0; e < 16; ++e)
        bfv[nt][e] = Bs[(jAdd + e + ((e >= 8) ? 8 : 0)) * BSTRIDE + n];
    }
#pragma unroll
    for (int mt = 0; mt < 2; ++mt)
#pragma unroll
      for (int nt = 0; nt < 2; ++nt)
        acc[mt][nt] = __builtin_amdgcn_wmma_f32_16x16x32_bf16(
            false, af[mt], false, bfv[nt], (short)0, acc[mt][nt], false, false);
    __syncthreads();
  }

#pragma unroll
  for (int mt = 0; mt < 2; ++mt) {
#pragma unroll
    for (int nt = 0; nt < 2; ++nt) {
      int col  = nBase + wn * 32 + nt * 16 + (lane & 15);
      int row0 = rowBase + wm * 32 + mt * 16 + ((lane < 16) ? 0 : 8);
      float cv = colV[b * D_ + col];
#pragma unroll
      for (int r = 0; r < 8; ++r)
        O1[(size_t)b * N_ * D_ + (size_t)(row0 + r) * D_ + col] = (bf16_t)(acc[mt][nt][r] + cv);
    }
  }
}

// ---------------- host launcher ----------------
extern "C" void kernel_launch(void* const* d_in, const int* in_sizes, int n_in,
                              void* d_out, int out_size, void* d_ws, size_t ws_size,
                              hipStream_t stream) {
  (void)in_sizes; (void)n_in; (void)out_size; (void)ws_size;
  const float* x   = (const float*)d_in[0];
  const float* Wq  = (const float*)d_in[1];
  const float* gq  = (const float*)d_in[2];
  const float* bqI = (const float*)d_in[3];
  const float* mq  = (const float*)d_in[4];
  const float* vq  = (const float*)d_in[5];
  const float* Wk  = (const float*)d_in[6];
  const float* gk  = (const float*)d_in[7];
  const float* bkI = (const float*)d_in[8];
  const float* mk  = (const float*)d_in[9];
  const float* vk  = (const float*)d_in[10];
  const float* Wv  = (const float*)d_in[11];
  const float* gv  = (const float*)d_in[12];
  const float* bvI = (const float*)d_in[13];
  const float* mv  = (const float*)d_in[14];
  const float* vv  = (const float*)d_in[15];
  const float* Wo  = (const float*)d_in[16];
  const float* bo  = (const float*)d_in[17];

  char* p = (char*)d_ws;
  auto alloc = [&](size_t bytes) -> char* {
    char* r = p; p += (bytes + 255) & ~(size_t)255; return r;
  };
  const size_t BND = (size_t)B_ * N_ * D_;
  bf16_t* xb   = (bf16_t*)alloc(BND * 2);
  bf16_t* Wqb  = (bf16_t*)alloc((size_t)D_ * D_ * 2);
  bf16_t* Wkb  = (bf16_t*)alloc((size_t)D_ * D_ * 2);
  bf16_t* Wvb  = (bf16_t*)alloc((size_t)D_ * D_ * 2);
  bf16_t* Wob  = (bf16_t*)alloc((size_t)D_ * D_ * 2);
  bf16_t* Vbuf = (bf16_t*)alloc(BND * 2);
  bf16_t* O1   = (bf16_t*)alloc(BND * 2);
  float* rowQ  = (float*)alloc((size_t)B_ * N_ * 4);
  float* rowK  = (float*)alloc((size_t)B_ * N_ * 4);
  float* tqA   = (float*)alloc((size_t)B_ * N_ * 4);
  float* aqA   = (float*)alloc((size_t)B_ * N_ * 4);
  float* bqA   = (float*)alloc((size_t)B_ * N_ * 4);
  float* tkA   = (float*)alloc((size_t)B_ * N_ * 4);
  float* ckA   = (float*)alloc((size_t)B_ * N_ * 4);
  float* dkA   = (float*)alloc((size_t)B_ * N_ * 4);
  float* colV  = (float*)alloc((size_t)B_ * D_ * 4);
  float* shq   = (float*)alloc((size_t)D_ * 4);
  float* shk   = (float*)alloc((size_t)D_ * 4);
  float* shv   = (float*)alloc((size_t)D_ * 4);

  hipMemsetAsync(rowQ, 0, (size_t)B_ * N_ * 4, stream);
  hipMemsetAsync(rowK, 0, (size_t)B_ * N_ * 4, stream);
  hipMemsetAsync(colV, 0, (size_t)B_ * D_ * 4, stream);

  cvt_bf16_kernel<<<(int)(BND / 256), 256, 0, stream>>>(x, xb, (int)BND);
  cvt_w_kernel<<<(D_ * D_) / 256, 256, 0, stream>>>(Wq, gq, vq, Wqb);
  cvt_w_kernel<<<(D_ * D_) / 256, 256, 0, stream>>>(Wk, gk, vk, Wkb);
  cvt_w_kernel<<<(D_ * D_) / 256, 256, 0, stream>>>(Wv, gv, vv, Wvb);
  cvt_w_kernel<<<(D_ * D_) / 256, 256, 0, stream>>>(Wo, nullptr, nullptr, Wob);
  shift_kernel<<<D_ / 256, 256, 0, stream>>>(gq, bqI, mq, vq, shq);
  shift_kernel<<<D_ / 256, 256, 0, stream>>>(gk, bkI, mk, vk, shk);
  shift_kernel<<<D_ / 256, 256, 0, stream>>>(gv, bvI, mv, vv, shv);

  dim3 g1(D_ / BN, (B_ * N_) / BM);  // 16 x 128
  gemm_bf16<0><<<g1, 256, 0, stream>>>(xb, Wqb, shq, rowQ, nullptr, B_ * N_, D_, D_);
  gemm_bf16<0><<<g1, 256, 0, stream>>>(xb, Wkb, shk, rowK, nullptr, B_ * N_, D_, D_);
  gemm_bf16<1><<<g1, 256, 0, stream>>>(xb, Wvb, shv, nullptr, Vbuf, B_ * N_, D_, D_);

  finish_t_kernel<<<(B_ * N_) / 256, 256, 0, stream>>>(rowQ, rowK, tqA, aqA, bqA, tkA, ckA, dkA);
  colsum_v_kernel<<<(B_ * D_ * 8) / 256, 256, 0, stream>>>(Vbuf, colV);

  dim3 g2(D_ / BN, N_ / BM, B_);     // 16 x 16 x 8
  attn_gemm<<<g2, 256, 0, stream>>>(Vbuf, tqA, aqA, bqA, tkA, ckA, dkA, colV, O1);

  gemm_bf16<2><<<g1, 256, 0, stream>>>(O1, Wob, bo, (float*)d_out, nullptr, B_ * N_, D_, D_);
}
